// S2VT_16776142258256
// MI455X (gfx1250) — matrix operand order repaired and checked by reading, not compiled, not run on hardware
//
#include <hip/hip_runtime.h>
#include <hip/hip_bf16.h>
#include <math.h>
#include <stdint.h>

#define USE_ASYNC_LDS 1

// ---------------- problem constants (from setup_inputs) ----------------
constexpr int NB   = 64;      // batch
constexpr int NT   = 64;      // encoder time steps
constexpr int NF   = 2048;    // input feature dim
constexpr int NH   = 512;     // hidden
constexpr int NE   = 512;     // embedding dim
constexpr int NV   = 32000;   // vocab
constexpr int N3H  = 3 * NH;  // 1536
constexpr int NEH  = NE + NH; // 1024
constexpr int NLEN = 16;      // max decode len

// ---------------- WMMA GEMM tile config ----------------
// Block tile 64(M) x 256(N) x 32(K), 256 threads = 8 waves.
// Wave tile 32x64 = 2x4 WMMA tiles -> 8 v_wmma per K-step per wave.
constexpr int TBM  = 64;
constexpr int TBN  = 256;
constexpr int TBK  = 32;
constexpr int LPAD = 8;
constexpr int LDSW = TBK + LPAD;   // LDS row pitch: 40 bf16 = 80 B (16B aligned, conflict-free)

typedef __bf16 bf16;
typedef __attribute__((ext_vector_type(16))) __bf16 v16bf;
typedef __attribute__((ext_vector_type(8)))  __bf16 v8bf;
typedef __attribute__((ext_vector_type(8)))  float  v8f;

// ---- CDNA5 async global->LDS copy (VGLOBAL GLOBAL_LOAD_ASYNC_TO_LDS_B128, ASYNCcnt) ----
__device__ __forceinline__ void async_load_b128(const bf16* gsrc, void* lds_dst)
{
    // low 32 bits of a generic __shared__ pointer == workgroup-relative LDS byte offset
    unsigned ldsoff = (unsigned)(uintptr_t)lds_dst;
    unsigned long long gaddr = (unsigned long long)(uintptr_t)gsrc;
    asm volatile("global_load_async_to_lds_b128 %0, %1, off"
                 :: "v"(ldsoff), "v"(gaddr)
                 : "memory");
}
__device__ __forceinline__ void wait_asynccnt0()
{
    asm volatile("s_wait_asynccnt 0x0" ::: "memory");
}

// ---------------- one-shot f32 -> bf16 conversion (weights / x) ----------------
__global__ __launch_bounds__(256)
void f32_to_bf16(const float* __restrict__ src, bf16* __restrict__ dst, long n)
{
    const long i = ((long)blockIdx.x * 256 + threadIdx.x) * 4;
    if (i + 3 < n) {
        const float4 v = *(const float4*)(src + i);
        dst[i]     = (bf16)v.x;
        dst[i + 1] = (bf16)v.y;
        dst[i + 2] = (bf16)v.z;
        dst[i + 3] = (bf16)v.w;
    } else {
        for (int u = 0; u < 4 && i + u < n; ++u) dst[i + u] = (bf16)src[i + u];
    }
}

// C[m, n] = bias[n] + sum_k A[m, k] * W[n, k]     (A, W already bf16)
// A: bf16 [M, K] row-stride lda ; W: bf16 [N, K] row-stride K ; C: f32 row-stride ldc
// M % 64 == 0, N % 256 == 0, K % 32 == 0 (true for every GEMM in this model).
__global__ __launch_bounds__(256)
void gemm_wmma_bf16(const bf16* __restrict__ A, int lda,
                    const bf16* __restrict__ W,
                    const float* __restrict__ bias,
                    float* __restrict__ C, long ldc,
                    int K)
{
    __shared__ bf16 As[2][TBM][LDSW];   // A tile, [m][k], double buffered
    __shared__ bf16 Bs[2][TBN][LDSW];   // W tile, [n][k], double buffered

    const int tid  = threadIdx.x;
    const int lane = tid & 31;
    const int wave = tid >> 5;
    const int wm   = (wave >> 2) * 32;  // wave M offset: 2 waves in M
    const int wn   = (wave & 3)  * 64;  // wave N offset: 4 waves in N
    const int half = lane >> 4;         // 0: lanes 0-15, 1: lanes 16-31
    const int l15  = lane & 15;

    const int m0 = blockIdx.y * TBM;
    const int n0 = blockIdx.x * TBN;

    // staging: thread moves one 16B chunk (8 bf16) of A and four of W per K-step
    const int srow = tid >> 2;          // 0..63
    const int scol = (tid & 3) * 8;     // 0,8,16,24

    // 8 accumulators (2 M-tiles x 4 N-tiles), initialized with bias (column-only)
    v8f acc[2][4];
#pragma unroll
    for (int ni = 0; ni < 4; ++ni) {
        const float bv = bias ? bias[n0 + wn + ni * 16 + l15] : 0.0f;
#pragma unroll
        for (int mi = 0; mi < 2; ++mi)
#pragma unroll
            for (int r = 0; r < 8; ++r) acc[mi][ni][r] = bv;
    }

    // ---- prologue: stage K-tile 0 into buffer 0 ----
#if USE_ASYNC_LDS
    async_load_b128(A + (size_t)(m0 + srow) * lda + scol, &As[0][srow][scol]);
#pragma unroll
    for (int j = 0; j < 4; ++j)
        async_load_b128(W + (size_t)(n0 + srow + 64 * j) * K + scol,
                        &Bs[0][srow + 64 * j][scol]);
    wait_asynccnt0();
#else
    *(v8bf*)&As[0][srow][scol] = *(const v8bf*)(A + (size_t)(m0 + srow) * lda + scol);
#pragma unroll
    for (int j = 0; j < 4; ++j)
        *(v8bf*)&Bs[0][srow + 64 * j][scol] =
            *(const v8bf*)(W + (size_t)(n0 + srow + 64 * j) * K + scol);
#endif
    __syncthreads();

    int cur = 0;
    for (int k0 = 0; k0 < K; k0 += TBK) {
        const bool has_next = (k0 + TBK) < K;
        const int  nxt      = cur ^ 1;

        // ---- kick off next-tile copies (run in background during WMMA burst) ----
#if USE_ASYNC_LDS
        if (has_next) {
            async_load_b128(A + (size_t)(m0 + srow) * lda + (k0 + TBK) + scol,
                            &As[nxt][srow][scol]);
#pragma unroll
            for (int j = 0; j < 4; ++j)
                async_load_b128(W + (size_t)(n0 + srow + 64 * j) * K + (k0 + TBK) + scol,
                                &Bs[nxt][srow + 64 * j][scol]);
            if (k0 + 2 * TBK < K)  // warm L2 one tile further ahead
                __builtin_prefetch(W + (size_t)(n0 + srow) * K + (k0 + 2 * TBK) + scol, 0, 1);
        }
#else
        v8bf areg;
        v8bf breg[4];
        if (has_next) {
            areg = *(const v8bf*)(A + (size_t)(m0 + srow) * lda + (k0 + TBK) + scol);
#pragma unroll
            for (int j = 0; j < 4; ++j)
                breg[j] = *(const v8bf*)(W + (size_t)(n0 + srow + 64 * j) * K + (k0 + TBK) + scol);
            if (k0 + 2 * TBK < K)
                __builtin_prefetch(W + (size_t)(n0 + srow) * K + (k0 + 2 * TBK) + scol, 0, 1);
        }
#endif

        // ---- fragments per CDNA5 16-bit WMMA lane layout (ISA 7.12.2) ----
        //  A 16x32: row M = lane&15; VGPR j holds K = (j>>2)*16 + half*8 + 2*(j&3) + {0,1}
        //  B 32x16: col N = lane&15; VGPR j holds K = (j>>2)*16 + half*8 + 2*(j&3) + {0,1}
        v16bf fa[2];
#pragma unroll
        for (int mi = 0; mi < 2; ++mi)
#pragma unroll
            for (int i = 0; i < 16; ++i) {
                const int j = i >> 1, bit = i & 1;
                const int ka = ((j >> 2) << 4) + (half << 3) + ((j & 3) << 1) + bit;
                fa[mi][i] = As[cur][wm + mi * 16 + l15][ka];
            }
#pragma unroll
        for (int ni = 0; ni < 4; ++ni) {
            v16bf fb;
#pragma unroll
            for (int i = 0; i < 16; ++i)
                fb[i] = Bs[cur][wn + ni * 16 + l15][((i >> 3) << 4) + (half << 3) + (i & 7)];
#pragma unroll
            for (int mi = 0; mi < 2; ++mi)
                acc[mi][ni] = __builtin_amdgcn_wmma_f32_16x16x32_bf16(
                    false, fa[mi], false, fb, (short)0, acc[mi][ni], false, false);
        }

        // ---- retire next-tile copies, swap buffers ----
#if USE_ASYNC_LDS
        if (has_next) wait_asynccnt0();
#else
        if (has_next) {
            *(v8bf*)&As[nxt][srow][scol] = areg;
#pragma unroll
            for (int j = 0; j < 4; ++j)
                *(v8bf*)&Bs[nxt][srow + 64 * j][scol] = breg[j];
        }
#endif
        __syncthreads();   // single barrier per K-step (ping-pong buffers)
        cur ^= 1;
    }

    // ---- epilogue: C/D layout VGPR r -> M = r + half*8, N = lane&15 ----
#pragma unroll
    for (int mi = 0; mi < 2; ++mi)
#pragma unroll
        for (int ni = 0; ni < 4; ++ni) {
            const int cn = n0 + wn + ni * 16 + l15;
#pragma unroll
            for (int r = 0; r < 8; ++r) {
                const int m = m0 + wm + mi * 16 + half * 8 + r;
                C[(size_t)m * ldc + cn] = acc[mi][ni][r];
            }
        }
}

// ---------------- fused GRU gate math ----------------
// r = sig(i_r + h_r); z = sig(i_z + h_z); n = tanh(i_n + r*h_n); h = (1-z)*n + z*h
// Also maintains the bf16 shadow of h consumed by the next recurrent GEMM.
__global__ __launch_bounds__(256)
void gru_gates(const float* __restrict__ gi_base, long gi_rstride, // [NB rows of 3H]
               const float* __restrict__ gh,                       // [NB, 3H]
               float* __restrict__ h,                              // [NB, H] in/out
               bf16*  __restrict__ h_bf,                           // [NB, H] bf16 shadow
               float* __restrict__ enc_out_t, long enc_rstride)    // optional
{
    const int e = blockIdx.x * 256 + threadIdx.x;
    if (e >= NB * NH) return;
    const int b = e >> 9;        // / NH
    const int j = e & (NH - 1);
    const float* gi = gi_base + (size_t)b * gi_rstride;
    const float* gr = gh      + (size_t)b * N3H;
    const float ir = gi[j],          hr = gr[j];
    const float iz = gi[NH + j],     hz = gr[NH + j];
    const float in_ = gi[2*NH + j],  hn = gr[2*NH + j];
    const float r = 1.0f / (1.0f + expf(-(ir + hr)));
    const float z = 1.0f / (1.0f + expf(-(iz + hz)));
    const float n = tanhf(in_ + r * hn);
    const float hv  = h[(size_t)b * NH + j];
    const float out = (1.0f - z) * n + z * hv;
    h[(size_t)b * NH + j]    = out;
    h_bf[(size_t)b * NH + j] = (bf16)out;
    if (enc_out_t) enc_out_t[(size_t)b * enc_rstride + j] = out;
}

// ---------------- attention + embedding -> rnn_in (bf16, GEMM-ready) ----------------
__global__ __launch_bounds__(256)
void attn_embed(const float* __restrict__ enc_out, // [B, T, H] f32
                const float* __restrict__ h,       // [B, H] f32
                const float* __restrict__ Emb,     // [V, E] f32
                const int*   __restrict__ tok,     // [B]
                bf16* __restrict__ rnn_in)         // [B, E+H] bf16
{
    __shared__ float s[NT];
    __shared__ float s_inv;
    const int b = blockIdx.x, tid = threadIdx.x;

    // scores[t] = dot(enc_out[b,t,:], h[b,:]) / sqrt(H); 4 lanes per t
    const int t = tid >> 2, q = tid & 3;
    const float* eo = enc_out + ((size_t)b * NT + t) * NH;
    const float* hb = h + (size_t)b * NH;
    float p = 0.0f;
    for (int j = q; j < NH; j += 4) p += eo[j] * hb[j];
    p += __shfl_xor(p, 1, 32);
    p += __shfl_xor(p, 2, 32);
    if (q == 0) s[t] = p * (1.0f / sqrtf((float)NH));
    __syncthreads();

    if (tid == 0) {
        float mx = s[0];
        for (int i = 1; i < NT; ++i) mx = fmaxf(mx, s[i]);
        float sum = 0.0f;
        for (int i = 0; i < NT; ++i) { const float e = expf(s[i] - mx); s[i] = e; sum += e; }
        s_inv = 1.0f / sum;
    }
    __syncthreads();
    const float inv = s_inv;

    // ctx -> rnn_in[b, E:]
    for (int j = tid; j < NH; j += 256) {
        float acc = 0.0f;
        for (int tt = 0; tt < NT; ++tt)
            acc += s[tt] * enc_out[((size_t)b * NT + tt) * NH + j];
        rnn_in[(size_t)b * NEH + NE + j] = (bf16)(acc * inv);
    }
    // embedding -> rnn_in[b, :E]
    const int tk = tok[b];
    for (int j = tid; j < NE; j += 256)
        rnn_in[(size_t)b * NEH + j] = (bf16)Emb[(size_t)tk * NE + j];
}

// ---------------- per-row argmax over the vocab ----------------
__global__ __launch_bounds__(256)
void argmax_k(const float* __restrict__ logits_base, long row_stride,
              int* __restrict__ tok,
              float* __restrict__ tok_out, long tok_out_stride)
{
    __shared__ float sv[256];
    __shared__ int   si[256];
    const int b = blockIdx.x, tid = threadIdx.x;
    const float* lg = logits_base + (size_t)b * row_stride;
    float best = -__builtin_inff();
    int   bi   = 0;
    for (int j = tid; j < NV; j += 256) {
        const float v = lg[j];
        if (v > best) { best = v; bi = j; }
    }
    sv[tid] = best; si[tid] = bi;
    __syncthreads();
    for (int s2 = 128; s2 > 0; s2 >>= 1) {
        if (tid < s2) {
            const float ov = sv[tid + s2]; const int oi = si[tid + s2];
            if (ov > sv[tid] || (ov == sv[tid] && oi < si[tid])) { sv[tid] = ov; si[tid] = oi; }
        }
        __syncthreads();
    }
    if (tid == 0) {
        tok[b] = si[0];
        tok_out[(size_t)b * tok_out_stride] = (float)si[0];
    }
}

// ---------------- init: h = 0 (f32 + bf16), tok = bos ----------------
__global__ void init_state(float* __restrict__ h, bf16* __restrict__ h_bf,
                           int* __restrict__ tok, const int* __restrict__ bos)
{
    const int i = blockIdx.x * 256 + threadIdx.x;
    if (i < NB * NH) { h[i] = 0.0f; h_bf[i] = (bf16)0.0f; }
    if (i < NB) tok[i] = bos[0];
}

static inline void cvt(const float* s, bf16* d, long n, hipStream_t st) {
    f32_to_bf16<<<(int)((n / 4 + 255) / 256), 256, 0, st>>>(s, d, n);
}

extern "C" void kernel_launch(void* const* d_in, const int* in_sizes, int n_in,
                              void* d_out, int out_size, void* d_ws, size_t ws_size,
                              hipStream_t stream) {
    (void)in_sizes; (void)n_in; (void)out_size; (void)ws_size;
    const float* x    = (const float*)d_in[0];
    const float* Emb  = (const float*)d_in[1];
    const float* Wi_e = (const float*)d_in[2];
    const float* Wh_e = (const float*)d_in[3];
    const float* bi_e = (const float*)d_in[4];
    const float* bh_e = (const float*)d_in[5];
    const float* Wi_d = (const float*)d_in[6];
    const float* Wh_d = (const float*)d_in[7];
    const float* bi_d = (const float*)d_in[8];
    const float* bh_d = (const float*)d_in[9];
    const float* Wo   = (const float*)d_in[10];
    const float* bo   = (const float*)d_in[11];
    const int*   bos  = (const int*)d_in[13];
    float* out = (float*)d_out;

    // ---- workspace carve-up: f32 state (~35 MB) + bf16 operand pool (~63 MB) ----
    float* Gx      = (float*)d_ws;                      // [B*T, 3H] input projections
    float* enc_out = Gx      + (size_t)NB * NT * N3H;   // [B, T, H]
    float* h       = enc_out + (size_t)NB * NT * NH;    // [B, H]
    float* gh      = h       + (size_t)NB * NH;         // [B, 3H]
    float* gi      = gh      + (size_t)NB * N3H;        // [B, 3H]
    int*   tok     = (int*)(gi + (size_t)NB * N3H);     // [B]

    bf16* x_bf    = (bf16*)(tok + 256);                 // [B*T, F]
    bf16* Wi_e_bf = x_bf    + (size_t)NB * NT * NF;     // [3H, F]
    bf16* Wh_e_bf = Wi_e_bf + (size_t)N3H * NF;         // [3H, H]
    bf16* Wi_d_bf = Wh_e_bf + (size_t)N3H * NH;         // [3H, E+H]
    bf16* Wh_d_bf = Wi_d_bf + (size_t)N3H * NEH;        // [3H, H]
    bf16* Wo_bf   = Wh_d_bf + (size_t)N3H * NH;         // [V, H]
    bf16* h_bf    = Wo_bf   + (size_t)NV * NH;          // [B, H]
    bf16* rnn_bf  = h_bf    + (size_t)NB * NH;          // [B, E+H]

    // ---- one-shot bf16 conversion of all GEMM operands (reused many times) ----
    cvt(x,    x_bf,    (long)NB * NT * NF, stream);
    cvt(Wi_e, Wi_e_bf, (long)N3H * NF,     stream);
    cvt(Wh_e, Wh_e_bf, (long)N3H * NH,     stream);
    cvt(Wi_d, Wi_d_bf, (long)N3H * NEH,    stream);
    cvt(Wh_d, Wh_d_bf, (long)N3H * NH,     stream);
    cvt(Wo,   Wo_bf,   (long)NV * NH,      stream);

    init_state<<<(NB * NH + 255) / 256, 256, 0, stream>>>(h, h_bf, tok, bos);

    // Phase 1: batched input projection Gx = x @ Wi_e^T + bi_e  (4096 x 1536, K=2048)
    gemm_wmma_bf16<<<dim3(N3H / TBN, (NB * NT) / TBM), 256, 0, stream>>>(
        x_bf, NF, Wi_e_bf, bi_e, Gx, N3H, NF);

    // Phase 2: encoder recurrence (sequential over T)
    for (int t = 0; t < NT; ++t) {
        gemm_wmma_bf16<<<dim3(N3H / TBN, NB / TBM), 256, 0, stream>>>(
            h_bf, NH, Wh_e_bf, bh_e, gh, N3H, NH);
        gru_gates<<<(NB * NH + 255) / 256, 256, 0, stream>>>(
            Gx + (size_t)t * N3H, (long)NT * N3H, gh, h, h_bf,
            enc_out + (size_t)t * NH, (long)NT * NH);
    }

    // Phase 3: greedy decode
    for (int step = 0; step < NLEN; ++step) {
        attn_embed<<<NB, 256, 0, stream>>>(enc_out, h, Emb, tok, rnn_bf);
        gemm_wmma_bf16<<<dim3(N3H / TBN, NB / TBM), 256, 0, stream>>>(
            rnn_bf, NEH, Wi_d_bf, bi_d, gi, N3H, NEH);
        gemm_wmma_bf16<<<dim3(N3H / TBN, NB / TBM), 256, 0, stream>>>(
            h_bf, NH, Wh_d_bf, bh_d, gh, N3H, NH);
        gru_gates<<<(NB * NH + 255) / 256, 256, 0, stream>>>(
            gi, (long)N3H, gh, h, h_bf, nullptr, 0);
        // logits[b, step, :] = h2 @ Wo^T + bo  (row b lives at b*NLEN*NV + step*NV)
        gemm_wmma_bf16<<<dim3(NV / TBN, NB / TBM), 256, 0, stream>>>(
            h_bf, NH, Wo_bf, bo, out + (size_t)step * NV, (long)NLEN * NV, NH);
        argmax_k<<<NB, 256, 0, stream>>>(
            out + (size_t)step * NV, (long)NLEN * NV,
            tok, out + (size_t)NB * NLEN * NV + step, (long)NLEN);
    }
}
